// TokenLabelHead_35811437314813
// MI455X (gfx1250) — matrix operand order repaired and checked by reading, not compiled, not run on hardware
//
#include <hip/hip_runtime.h>
#include <hip/hip_bf16.h>
#include <cstdint>

// Problem constants (from reference)
#define B_    32
#define L_    1024
#define DIN_  2048
#define H_    1024
#define NL_   21
#define NLP_  32      // NL padded to 2 WMMA N-tiles
#define E_    768
#define LN_EPS 1e-5f
#define MT_   4       // M-tiles per wave (M_block = 16*MT_)

typedef __attribute__((ext_vector_type(16))) __bf16 v16bf;
typedef __attribute__((ext_vector_type(8)))  __bf16 v8bf;
typedef __attribute__((ext_vector_type(8)))  float  v8f;

__device__ inline v8f wmma_bf16(v16bf a, v16bf b, v8f c) {
  // 8 args: (neg_a, A, neg_b, B, c_mod, C, reuse_a, reuse_b)
  return __builtin_amdgcn_wmma_f32_16x16x32_bf16(
      false, a, false, b, (short)0, c, false, false);
}

// Combine two 16-byte LDS runs into one fragment (register-coalesced).
__device__ inline v16bf cat8(v8bf lo, v8bf hi) {
  v16bf r;
  #pragma unroll
  for (int e = 0; e < 8; ++e) { r[e] = lo[e]; r[8 + e] = hi[e]; }
  return r;
}

// ---------------------------------------------------------------------------
// Stage 0: W_tok fp32 -> bf16 copy (once; 4 MB, lives in L2 afterwards).
// ---------------------------------------------------------------------------
__global__ void wtok_bf16_kernel(const float* __restrict__ Wt,
                                 __bf16* __restrict__ Wtb) {
  const size_t i = ((size_t)blockIdx.x * blockDim.x + threadIdx.x) * 4;
  const float4 f = *(const float4*)(Wt + i);
  Wtb[i + 0] = (__bf16)f.x;
  Wtb[i + 1] = (__bf16)f.y;
  Wtb[i + 2] = (__bf16)f.z;
  Wtb[i + 3] = (__bf16)f.w;
}

// ---------------------------------------------------------------------------
// Stage 1: label_hT[k][n] = (label_table @ W_lab)^T, bf16, N padded to 32.
// ---------------------------------------------------------------------------
__global__ void label_head_kernel(const float* __restrict__ label_table,
                                  const float* __restrict__ W_lab,
                                  __bf16* __restrict__ labT) {
  int idx = blockIdx.x * blockDim.x + threadIdx.x;   // H_*NLP_ threads
  int k = idx >> 5;        // H index
  int n = idx & 31;        // padded label index
  float acc = 0.f;
  if (n < NL_) {
    const float* lt = label_table + (size_t)n * E_;
    const float* wl = W_lab + k;                     // W_lab[e*H + k]
    for (int e = 0; e < E_; ++e) acc += lt[e] * wl[(size_t)e * H_];
  }
  labT[(size_t)k * NLP_ + n] = (__bf16)acc;
}

// ---------------------------------------------------------------------------
// Stage 2: fused  seq@W_tok -> LayerNorm -> GELU(exact) -> @label_h^T * temp
// One workgroup (8 wave32) per 64-token tile (MT_=4 WMMA M-tiles per wave).
// Wave w owns H columns [w*128, w*128+128) as 8 WMMA N-tiles, fp32 accum
// (acc = 4*8 v8f = 256 VGPRs; gfx1250 VGPR-MSB gives 1024-reg headroom).
// Each B fragment is loaded once and feeds 4 WMMAs -> W_tok L2 traffic
// drops to 4MB * (B*L/64) = 2 GB total.
// ---------------------------------------------------------------------------
__global__ __launch_bounds__(256)
void fused_token_head_kernel(const float*  __restrict__ seq,
                             const __bf16* __restrict__ Wtb,
                             const float*  __restrict__ gamma,
                             const float*  __restrict__ beta,
                             const __bf16* __restrict__ labT,
                             const float*  __restrict__ temp_p,
                             float*        __restrict__ out_logits) {
  __shared__ __bf16 sA[2][16 * MT_][32];  // double-buffered 64x32 seq tile
  __shared__ __bf16 sTok[16][H_];         // post-GELU token_h, per M-tile
  __shared__ float  sRed[2][8][16];       // per-wave row sum / sumsq
  __shared__ float  sMean[16], sRstd[16];
  __shared__ float  sLogit[16][NLP_];     // logits accumulator (fp32)

  const int tid  = threadIdx.x;
  const int wave = tid >> 5;
  const int lane = tid & 31;
  const int half = lane >> 4;   // C/D layout: lanes 16-31 hold rows M+8
  const int l16  = lane & 15;
  const size_t t0 = (size_t)blockIdx.x * (16 * MT_);  // first token row

  v8f acc[MT_][8];
  #pragma unroll
  for (int mt = 0; mt < MT_; ++mt)
    #pragma unroll
    for (int i = 0; i < 8; ++i) acc[mt][i] = {};

  // cooperative staging: 64x32 fp32 seq tile -> bf16 LDS (8 elems/thread)
  auto stageA = [&](int kt) {
    const int buf = kt & 1, k0 = kt * 32;
    const int j = tid * 8, m = j >> 5, kk = j & 31;
    const float4 f0 = *(const float4*)(seq + (t0 + m) * DIN_ + k0 + kk);
    const float4 f1 = *(const float4*)(seq + (t0 + m) * DIN_ + k0 + kk + 4);
    sA[buf][m][kk]     = (__bf16)f0.x;
    sA[buf][m][kk + 1] = (__bf16)f0.y;
    sA[buf][m][kk + 2] = (__bf16)f0.z;
    sA[buf][m][kk + 3] = (__bf16)f0.w;
    sA[buf][m][kk + 4] = (__bf16)f1.x;
    sA[buf][m][kk + 5] = (__bf16)f1.y;
    sA[buf][m][kk + 6] = (__bf16)f1.z;
    sA[buf][m][kk + 7] = (__bf16)f1.w;
  };

  stageA(0);
  const int n0w = wave * 128;

  for (int kt = 0; kt < 64; ++kt) {               // K = 2048 in steps of 32
    __syncthreads();
    if (kt + 1 < 64) stageA(kt + 1);              // fill other buffer
    const int buf = kt & 1, k0 = kt * 32;

    // MT_ A fragments (rows mt*16 .. mt*16+15), ISA 7.12.2 lane-half split
    v16bf a[MT_];
    #pragma unroll
    for (int mt = 0; mt < MT_; ++mt)
      a[mt] = cat8(*(const v8bf*)(&sA[buf][mt * 16 + l16][half * 8]),
                   *(const v8bf*)(&sA[buf][mt * 16 + l16][16 + half * 8]));

    const __bf16* bbase = Wtb + (size_t)k0 * H_ + n0w + l16;
    // pull next K-step's B row toward the caches (global_prefetch_b8)
    __builtin_prefetch(bbase + 32 * H_, 0, 1);
    #pragma unroll
    for (int nt = 0; nt < 8; ++nt) {
      v16bf b;   // column gather, table k order
      #pragma unroll
      for (int e = 0; e < 16; ++e) {
        const int kb = (e < 8) ? (half * 8 + e) : (16 + half * 8 + (e - 8));
        b[e] = bbase[(size_t)kb * H_ + nt * 16];
      }
      #pragma unroll
      for (int mt = 0; mt < MT_; ++mt)
        acc[mt][nt] = wmma_bf16(a[mt], b, acc[mt][nt]);
    }
  }

  const float temp = temp_p[0];

  // ---- epilogue per M-tile: LN -> GELU -> logits GEMM -> write ----
  #pragma unroll
  for (int mt = 0; mt < MT_; ++mt) {
    // LayerNorm statistics: butterfly within 16-lane halves
    float rs[8], rq[8];
    #pragma unroll
    for (int r = 0; r < 8; ++r) { rs[r] = 0.f; rq[r] = 0.f; }
    #pragma unroll
    for (int nt = 0; nt < 8; ++nt) {
      #pragma unroll
      for (int r = 0; r < 8; ++r) {
        float s = acc[mt][nt][r], q = s * s;
        #pragma unroll
        for (int m = 8; m >= 1; m >>= 1) {
          s += __shfl_xor(s, m, 32);
          q += __shfl_xor(q, m, 32);
        }
        rs[r] += s; rq[r] += q;   // row (half*8 + r), this wave's 128 cols
      }
    }
    if (l16 == 0) {
      #pragma unroll
      for (int r = 0; r < 8; ++r) {
        sRed[0][wave][half * 8 + r] = rs[r];
        sRed[1][wave][half * 8 + r] = rq[r];
      }
    }
    for (int i = tid; i < 16 * NLP_; i += 256) (&sLogit[0][0])[i] = 0.f;
    __syncthreads();

    if (tid < 16) {
      float s = 0.f, q = 0.f;
      #pragma unroll
      for (int w = 0; w < 8; ++w) { s += sRed[0][w][tid]; q += sRed[1][w][tid]; }
      const float mean = s * (1.f / H_);
      const float var  = q * (1.f / H_) - mean * mean;  // ddof=0 (jnp.var)
      sMean[tid] = mean;
      sRstd[tid] = rsqrtf(var + LN_EPS);
    }
    __syncthreads();

    // normalize + exact GELU, write token_h (bf16) tile into LDS
    #pragma unroll
    for (int nt = 0; nt < 8; ++nt) {
      const int n = n0w + nt * 16 + l16;
      const float g = gamma[n], bt = beta[n];
      #pragma unroll
      for (int r = 0; r < 8; ++r) {
        const int m = half * 8 + r;
        const float y  = (acc[mt][nt][r] - sMean[m]) * sRstd[m] * g + bt;
        const float gl = 0.5f * y * (1.f + erff(y * 0.70710678118654752f));
        sTok[m][n] = (__bf16)gl;
      }
    }
    __syncthreads();

    // logits GEMM: K=1024 split over 8 waves (128 each), N = 2 tiles
    v8f c0 = {}, c1 = {};
    #pragma unroll
    for (int kk = 0; kk < 4; ++kk) {
      const int k0 = wave * 128 + kk * 32;
      const v16bf a =
          cat8(*(const v8bf*)(&sTok[l16][k0 + half * 8]),
               *(const v8bf*)(&sTok[l16][k0 + 16 + half * 8]));
      const __bf16* bp = labT + (size_t)k0 * NLP_ + l16;
      v16bf b0, b1;
      #pragma unroll
      for (int e = 0; e < 16; ++e) {
        const int kb = (e < 8) ? (half * 8 + e) : (16 + half * 8 + (e - 8));
        b0[e] = bp[(size_t)kb * NLP_];
        b1[e] = bp[(size_t)kb * NLP_ + 16];
      }
      c0 = wmma_bf16(a, b0, c0);
      c1 = wmma_bf16(a, b1, c1);
    }
    #pragma unroll
    for (int r = 0; r < 8; ++r) {
      const int m = half * 8 + r;
      atomicAdd(&sLogit[m][l16],      c0[r]);   // ds_add_f32
      atomicAdd(&sLogit[m][16 + l16], c1[r]);
    }
    __syncthreads();

    for (int i = tid; i < 16 * NL_; i += 256) {
      const int m = i / NL_, j = i % NL_;
      out_logits[(t0 + mt * 16 + m) * NL_ + j] = sLogit[m][j] * temp;
    }
    __syncthreads();   // before next M-tile reuses sRed/sTok/sLogit
  }
}

// ---------------------------------------------------------------------------
// Stage 3: CRF negative log-likelihood. One wave32 per batch; lane j = tag j.
// ---------------------------------------------------------------------------
__global__ void crf_kernel(const float* __restrict__ logits,
                           const int* __restrict__ labels,
                           const unsigned char* __restrict__ mask,
                           const float* __restrict__ start_t,
                           const float* __restrict__ end_t,
                           const float* __restrict__ trans,
                           float* __restrict__ part) {
  const int b = blockIdx.x;
  const int j = threadIdx.x;                // 32 lanes, tags 0..20 valid
  const float NEG = -1e30f;

  float tcol[NL_];                           // trans[:, j]
  #pragma unroll 1
  for (int i = 0; i < NL_; ++i) tcol[i] = (j < NL_) ? trans[i * NL_ + j] : 0.f;

  const float* lg = logits + (size_t)b * L_ * NL_;
  float alpha = (j < NL_) ? (start_t[j] + lg[j]) : NEG;

  for (int t = 1; t < L_; ++t) {
    float m = NEG, s = 0.f;
    #pragma unroll 1
    for (int i = 0; i < NL_; ++i) {
      const float v  = __shfl(alpha, i, 32) + tcol[i];
      const float mn = fmaxf(m, v);
      s = s * __expf(m - mn) + __expf(v - mn);
      m = mn;
    }
    const float e  = (j < NL_) ? lg[(size_t)t * NL_ + j] : 0.f;
    const float na = m + __logf(s) + e;
    if (j < NL_ && mask[b * L_ + t]) alpha = na;
  }

  // logZ = logsumexp(alpha + end_trans)
  float v = (j < NL_) ? (alpha + end_t[j]) : NEG;
  float M = v;
  #pragma unroll
  for (int d = 16; d >= 1; d >>= 1) M = fmaxf(M, __shfl_xor(M, d, 32));
  float s = __expf(v - M);
  #pragma unroll
  for (int d = 16; d >= 1; d >>= 1) s += __shfl_xor(s, d, 32);
  const float logZ = M + __logf(s);

  // gold-path score, strided over lanes
  const int* lab = labels + b * L_;
  const unsigned char* mk = mask + b * L_;
  float sc = 0.f; int cnt = 0;
  for (int t = j; t < L_; t += 32) {
    const int yt = lab[t];
    cnt += (mk[t] != 0);
    if (t == 0) {
      sc += start_t[yt] + lg[yt];
    } else {
      const float mf = (mk[t] != 0) ? 1.f : 0.f;
      sc += mf * (lg[(size_t)t * NL_ + yt] + trans[lab[t - 1] * NL_ + yt]);
    }
  }
  #pragma unroll
  for (int d = 16; d >= 1; d >>= 1) {
    sc  += __shfl_xor(sc, d, 32);
    cnt += __shfl_xor(cnt, d, 32);
  }
  if (j == 0) {
    const int last = cnt - 1;
    sc += end_t[lab[last]];
    part[b] = sc - logZ;
  }
}

__global__ void loss_kernel(const float* __restrict__ part,
                            float* __restrict__ out_loss) {
  float v = part[threadIdx.x];
  #pragma unroll
  for (int d = 16; d >= 1; d >>= 1) v += __shfl_xor(v, d, 32);
  if (threadIdx.x == 0) out_loss[0] = -(v * (1.f / B_));
}

// ---------------------------------------------------------------------------
extern "C" void kernel_launch(void* const* d_in, const int* in_sizes, int n_in,
                              void* d_out, int out_size, void* d_ws, size_t ws_size,
                              hipStream_t stream) {
  (void)in_sizes; (void)n_in; (void)out_size; (void)ws_size;
  const float* seq    = (const float*)d_in[0];
  const int*   labels = (const int*)d_in[1];
  const unsigned char* mask = (const unsigned char*)d_in[2];
  const float* Wt     = (const float*)d_in[3];
  const float* gamma  = (const float*)d_in[4];
  const float* beta   = (const float*)d_in[5];
  const float* ltab   = (const float*)d_in[6];
  const float* Wlab   = (const float*)d_in[7];
  const float* temp   = (const float*)d_in[8];
  const float* st     = (const float*)d_in[9];
  const float* et     = (const float*)d_in[10];
  const float* tr     = (const float*)d_in[11];

  float* out_logits = (float*)d_out;
  float* out_loss   = out_logits + (size_t)B_ * L_ * NL_;

  // workspace layout: [ Wtb bf16 4MB | labT bf16 64KB | part 128B ]
  __bf16* Wtb  = (__bf16*)d_ws;
  __bf16* labT = (__bf16*)((char*)d_ws + (size_t)DIN_ * H_ * 2);
  float*  part = (float*)((char*)d_ws + (size_t)DIN_ * H_ * 2
                                      + (size_t)H_ * NLP_ * 2 + 256);

  hipLaunchKernelGGL(wtok_bf16_kernel, dim3((DIN_ * H_) / (256 * 4)), dim3(256),
                     0, stream, Wt, Wtb);
  hipLaunchKernelGGL(label_head_kernel, dim3((H_ * NLP_) / 256), dim3(256),
                     0, stream, ltab, Wlab, labT);
  hipLaunchKernelGGL(fused_token_head_kernel, dim3((B_ * L_) / (16 * MT_)),
                     dim3(256), 0, stream, seq, Wtb, gamma, beta, labT, temp,
                     out_logits);
  hipLaunchKernelGGL(crf_kernel, dim3(B_), dim3(32),
                     0, stream, out_logits, labels, mask, st, et, tr, part);
  hipLaunchKernelGGL(loss_kernel, dim3(1), dim3(32),
                     0, stream, part, out_loss);
}
